// Conv1x1_ssa_60335700574212
// MI455X (gfx1250) — matrix operand order repaired and checked by reading, not compiled, not run on hardware
//
#include <hip/hip_runtime.h>

// ---- problem dims (fixed by reference) ----
#define T_   4
#define B_   16
#define C_   256
#define O_   256
#define HW_  1024          // 32*32
#define TB_  64            // T_*B_
#define NSTAT 65536.0f     // TB_*HW_ samples per channel for BN

typedef __attribute__((ext_vector_type(16))) __bf16 v16bf;
typedef __attribute__((ext_vector_type(8)))  __bf16 v8bf;
typedef __attribute__((ext_vector_type(8)))  float  v8f;
typedef int v4i __attribute__((vector_size(16)));   // pointee type expected by async-to-LDS builtin

#if __has_builtin(__builtin_amdgcn_global_load_async_to_lds_b128)
#define HAVE_ASYNC_LDS 1
#else
#define HAVE_ASYNC_LDS 0
#endif

__device__ __forceinline__ unsigned short f32_to_bf16_rne(float f) {
    unsigned u = __float_as_uint(f);
    unsigned r = 0x7FFFu + ((u >> 16) & 1u);
    return (unsigned short)((u + r) >> 16);
}

// ---------------- init: zero BN accumulators + absmax ----------------
__global__ void init_zero_kernel(unsigned* p, int n) {
    int i = blockIdx.x * 256 + threadIdx.x;
    if (i < n) p[i] = 0u;
}

// ---------------- max |w| reduction ----------------
__global__ void absmax_kernel(const float* __restrict__ w, unsigned* __restrict__ absmax, int n) {
    __shared__ float red[256];
    float m = 0.f;
    for (int i = blockIdx.x * 256 + threadIdx.x; i < n; i += gridDim.x * 256)
        m = fmaxf(m, fabsf(w[i]));
    red[threadIdx.x] = m;
    __syncthreads();
    for (int s = 128; s > 0; s >>= 1) {
        if (threadIdx.x < s) red[threadIdx.x] = fmaxf(red[threadIdx.x], red[threadIdx.x + s]);
        __syncthreads();
    }
    if (threadIdx.x == 0) atomicMax(absmax, __float_as_uint(red[0]));  // non-neg floats: uint order == float order
}

// ---------------- fake-quant: store integer q in bf16 (exact), scale separately ----------------
__global__ void quant_kernel(const float* __restrict__ w, const unsigned* __restrict__ absmax,
                             unsigned short* __restrict__ wq, float* __restrict__ scale_out, int n) {
    float scale = __uint_as_float(*absmax) * (1.0f / 127.0f);
    int i = blockIdx.x * 256 + threadIdx.x;
    if (i < n) {
        float q = rintf(w[i] / scale);
        q = fminf(fmaxf(q, -127.f), 127.f);
        wq[i] = f32_to_bf16_rne(q);   // |q| <= 127 -> exact in bf16
    }
    if (i == 0) *scale_out = scale;
}

// ---------------- x[tb][c][hw] (f32) -> xT[tb][hw][c] (bf16) via LDS tile transpose ----------------
__global__ void __launch_bounds__(256)
transpose_kernel(const float* __restrict__ x, unsigned short* __restrict__ xT) {
    __shared__ float tile[32][33];
    const int hw0 = blockIdx.x * 32;
    const int c0  = blockIdx.y * 32;
    const int tb  = blockIdx.z;
    #pragma unroll
    for (int j = 0; j < 4; ++j) {
        int c = c0 + threadIdx.y + 8 * j;
        tile[threadIdx.y + 8 * j][threadIdx.x] =
            x[((size_t)tb * C_ + c) * HW_ + hw0 + threadIdx.x];
    }
    __syncthreads();
    #pragma unroll
    for (int j = 0; j < 4; ++j) {
        int hw = hw0 + threadIdx.y + 8 * j;
        xT[((size_t)tb * HW_ + hw) * C_ + c0 + threadIdx.x] =
            f32_to_bf16_rne(tile[threadIdx.x][threadIdx.y + 8 * j]);
    }
}

// ---------------- main GEMM: y[tb][o][hw] = scale * (q @ xT), + BN sum/sumsq ----------------
// Grid (HW_/128, O_/128, TB_) = (8,2,64); 256 threads = 8 wave32; each wave: 4(M)x2(N) 16x16 tiles.
// The 128x256 bf16 weight tile (64 KB) is staged once per workgroup into LDS via
// GLOBAL_LOAD_ASYNC_TO_LDS_B128 (ASYNCcnt) and reused by all 8 waves from ds_load_b128.
__global__ void __launch_bounds__(256)
gemm_bn_kernel(const unsigned short* __restrict__ wq_u16,
               const unsigned short* __restrict__ xT_u16,
               const float* __restrict__ scale_p,
               float* __restrict__ y,
               float* __restrict__ gSum, float* __restrict__ gSq) {
    __shared__ unsigned short ldsA[128 * C_];   // 64 KB weight tile
    __shared__ float sSum[128];
    __shared__ float sSq[128];

    const int lane  = threadIdx.x & 31;
    const int wave  = threadIdx.x >> 5;
    const int half  = lane >> 4;     // lane 0-15 vs 16-31
    const int l15   = lane & 15;
    const int tb    = blockIdx.z;
    const int oBase = blockIdx.y * 128;
    const int nBase = blockIdx.x * 128;
    const int oLoc0 = (wave >> 2) * 64;           // wave's M origin within the LDS tile
    const int n0    = nBase + (wave & 3) * 32;    // 4 waves in N

    if (threadIdx.x < 128) { sSum[threadIdx.x] = 0.f; sSq[threadIdx.x] = 0.f; }

    // ---- stage A tile: 4096 x b128 chunks, 16 per thread, coalesced ----
    {
        const unsigned short* gA = wq_u16 + (size_t)oBase * C_;
        #pragma unroll
        for (int j = 0; j < 16; ++j) {
            const int chunk = threadIdx.x + 256 * j;   // 16B chunk index
            const int elem  = chunk * 8;               // 8 bf16 per chunk
#if HAVE_ASYNC_LDS
            auto gsrc = (__attribute__((address_space(1))) v4i*)
                        (unsigned long long)(size_t)(const void*)(gA + elem);
            auto ldst = (__attribute__((address_space(3))) v4i*)
                        (unsigned)(size_t)(void*)&ldsA[elem];
            __builtin_amdgcn_global_load_async_to_lds_b128(gsrc, ldst, 0, 0);
#else
            *(uint4*)&ldsA[elem] = *(const uint4*)(gA + elem);   // global_load_b128 + ds_store_b128
#endif
        }
#if HAVE_ASYNC_LDS
#if __has_builtin(__builtin_amdgcn_s_wait_asynccnt)
        __builtin_amdgcn_s_wait_asynccnt(0);
#else
        asm volatile("s_wait_asynccnt 0" ::: "memory");
#endif
#endif
    }
    __syncthreads();

    const __bf16* lA  = (const __bf16*)ldsA;
    const __bf16* xTb = (const __bf16*)xT_u16;

    v8f acc[4][2];
    #pragma unroll
    for (int mt = 0; mt < 4; ++mt)
        #pragma unroll
        for (int nt = 0; nt < 2; ++nt)
            acc[mt][nt] = v8f{0.f, 0.f, 0.f, 0.f, 0.f, 0.f, 0.f, 0.f};

    for (int k0 = 0; k0 < C_; k0 += 32) {
        // A fragments from LDS: 16x32 bf16 layout — lane<16: K {k0..k0+7, k0+16..k0+23}; lane>=16: +8
        v16bf afr[4];
        #pragma unroll
        for (int mt = 0; mt < 4; ++mt) {
            const __bf16* p = lA + (size_t)(oLoc0 + mt * 16 + l15) * C_ + k0 + half * 8;
            v8bf lo = *(const v8bf*)p;           // ds_load_b128
            v8bf hi = *(const v8bf*)(p + 16);    // ds_load_b128
            #pragma unroll
            for (int e = 0; e < 8; ++e) { afr[mt][e] = lo[e]; afr[mt][e + 8] = hi[e]; }
        }
        // B fragments from global (L2-resident xT): lane<16: col=l15, K {k0..k0+7, k0+16..k0+23}; lane>=16: +8
        v16bf bfr[2];
        #pragma unroll
        for (int nt = 0; nt < 2; ++nt) {
            const __bf16* p = xTb + ((size_t)tb * HW_ + n0 + nt * 16 + l15) * C_ + k0 + half * 8;
            v8bf lo = *(const v8bf*)p;
            v8bf hi = *(const v8bf*)(p + 16);
            #pragma unroll
            for (int e = 0; e < 8; ++e) { bfr[nt][e] = lo[e]; bfr[nt][e + 8] = hi[e]; }
            if (k0 + 32 < C_) __builtin_prefetch(p + 32, 0, 3);  // global_prefetch_b8, near-temporal
        }
        #pragma unroll
        for (int mt = 0; mt < 4; ++mt)
            #pragma unroll
            for (int nt = 0; nt < 2; ++nt)
                acc[mt][nt] = __builtin_amdgcn_wmma_f32_16x16x32_bf16(
                    false, afr[mt], false, bfr[nt], (short)0, acc[mt][nt], false, false);
    }

    const float scale = *scale_p;
    // D layout: VGPR r, lane l -> row = r + 8*(l>>4), col = l&15
    #pragma unroll
    for (int mt = 0; mt < 4; ++mt) {
        #pragma unroll
        for (int nt = 0; nt < 2; ++nt) {
            #pragma unroll
            for (int r = 0; r < 8; ++r) {
                const int oL = oLoc0 + mt * 16 + r + 8 * half;   // channel within tile
                const int o  = oBase + oL;
                const int n  = n0 + nt * 16 + l15;
                float val = acc[mt][nt][r] * scale;
                y[((size_t)tb * O_ + o) * HW_ + n] = val;
                // reduce over the 16 lanes of this half-wave (same channel o, distinct n)
                float s1 = val, s2 = val * val;
                #pragma unroll
                for (int m = 1; m < 16; m <<= 1) {
                    s1 += __shfl_xor(s1, m, 32);
                    s2 += __shfl_xor(s2, m, 32);
                }
                if (l15 == 0) {
                    atomicAdd(&sSum[oL], s1);   // ds_add_f32
                    atomicAdd(&sSq [oL], s2);
                }
            }
        }
    }
    __syncthreads();
    if (threadIdx.x < 128) {
        atomicAdd(&gSum[oBase + threadIdx.x], sSum[threadIdx.x]);  // global_atomic_add_f32
        atomicAdd(&gSq [oBase + threadIdx.x], sSq [threadIdx.x]);
    }
}

// ---------------- BN finalize: per-channel inv/shift ----------------
__global__ void bn_finalize_kernel(const float* __restrict__ gSum, const float* __restrict__ gSq,
                                   const float* __restrict__ gamma, const float* __restrict__ beta,
                                   float* __restrict__ invP, float* __restrict__ shiftP) {
    int o = threadIdx.x;
    float mean = gSum[o] * (1.0f / NSTAT);
    float var  = gSq[o] * (1.0f / NSTAT) - mean * mean;   // biased, as torch BN forward
    float inv  = gamma[o] * rsqrtf(var + 1e-5f);
    invP[o]   = inv;
    shiftP[o] = beta[o] - mean * inv;
}

// ---------------- fused BN affine + LIF over T ----------------
__global__ void __launch_bounds__(256)
lif_kernel(const float* __restrict__ y, const float* __restrict__ invP,
           const float* __restrict__ shiftP, float* __restrict__ out) {
    const int idx = blockIdx.x * 256 + threadIdx.x;      // over B_*O_*HW_
    const int hw = idx & (HW_ - 1);
    const int o  = (idx >> 10) & (O_ - 1);
    const int b  = idx >> 18;
    const float inv = invP[o], sh = shiftP[o];
    float v = 0.f;
    #pragma unroll
    for (int t = 0; t < T_; ++t) {
        size_t off = (((size_t)(t * B_ + b) * O_ + o) * HW_) + hw;
        float yv = y[off] * inv + sh;                    // BN
        v += (yv - v) * 0.5f;                            // tau = 2
        float s = (v >= 1.0f) ? 1.0f : 0.0f;             // heaviside(v - v_th)
        v *= (1.0f - s);                                 // hard reset
        out[off] = s;
    }
}

// ---------------- host launch ----------------
extern "C" void kernel_launch(void* const* d_in, const int* in_sizes, int n_in,
                              void* d_out, int out_size, void* d_ws, size_t ws_size,
                              hipStream_t stream) {
    const float* x     = (const float*)d_in[0];   // (4,16,256,32,32)
    const float* w     = (const float*)d_in[1];   // (256,256)
    const float* gamma = (const float*)d_in[2];   // (256,)
    const float* beta  = (const float*)d_in[3];   // (256,)
    float* out = (float*)d_out;                   // (4,16,256,32,32) f32 spikes

    // workspace carve-up (~101 MB total)
    char* ws = (char*)d_ws;
    float*          y     = (float*)ws;                                   //  67108864 B
    unsigned short* xT    = (unsigned short*)(ws + 67108864);             //  33554432 B
    unsigned short* wq    = (unsigned short*)(ws + 67108864 + 33554432);  //    131072 B
    float* stats  = (float*)(ws + 100663296 + 131072);
    float*    gSum    = stats;            // 256
    float*    gSq     = stats + 256;      // 256
    unsigned* absmaxP = (unsigned*)(stats + 512); // 1
    float*    scaleP  = stats + 513;      // 1
    float*    invP    = stats + 514;      // 256
    float*    shiftP  = stats + 770;      // 256

    // 1) zero gSum/gSq/absmax (513 words)
    init_zero_kernel<<<3, 256, 0, stream>>>((unsigned*)stats, 513);
    // 2) scale = max|w| / 127
    absmax_kernel<<<64, 256, 0, stream>>>(w, absmaxP, O_ * C_);
    // 3) wq = bf16(round-clip(w/scale))  (exact integers in bf16)
    quant_kernel<<<(O_ * C_ + 255) / 256, 256, 0, stream>>>(w, absmaxP, wq, scaleP, O_ * C_);
    // 4) xT[tb][hw][c] = bf16(x[tb][c][hw])
    transpose_kernel<<<dim3(HW_ / 32, C_ / 32, TB_), dim3(32, 8), 0, stream>>>(x, xT);
    // 5) WMMA GEMM (LDS-staged A via async-to-LDS) + BN statistics
    gemm_bn_kernel<<<dim3(HW_ / 128, O_ / 128, TB_), 256, 0, stream>>>(wq, xT, scaleP, y, gSum, gSq);
    // 6) BN inv/shift
    bn_finalize_kernel<<<1, 256, 0, stream>>>(gSum, gSq, gamma, beta, invP, shiftP);
    // 7) fused BN + LIF, write spikes
    lif_kernel<<<(B_ * O_ * HW_) / 256, 256, 0, stream>>>(y, invP, shiftP, out);
}
